// SemiSelfAttention_56538949484918
// MI455X (gfx1250) — matrix operand order repaired and checked by reading, not compiled, not run on hardware
//
#include <hip/hip_runtime.h>
#include <hip/hip_bf16.h>

typedef __bf16 bf16_t;
typedef __attribute__((ext_vector_type(16))) __bf16 v16bf;
typedef __attribute__((ext_vector_type(8)))  float  v8f;

#if __has_builtin(__builtin_amdgcn_global_load_async_to_lds_b128) && \
    __has_builtin(__builtin_amdgcn_s_wait_asynccnt)
#define HAVE_ASYNC 1
// Builtin signature (from hipcc diagnostics): (int4 AS1*, int4 AS3*, int, int)
typedef int v4i_vec __attribute__((__vector_size__(16)));
typedef __attribute__((address_space(1))) v4i_vec as1_v4i;
typedef __attribute__((address_space(3))) v4i_vec as3_v4i;
#define ASYNC_LDS_B128(gsrc, ldst)                                         \
    __builtin_amdgcn_global_load_async_to_lds_b128(                        \
        (as1_v4i*)(gsrc), (as3_v4i*)(ldst), 0, 0)
#define WAIT_ASYNC() __builtin_amdgcn_s_wait_asynccnt(0)
#else
#define HAVE_ASYNC 0
#define WAIT_ASYNC()
#endif

// ---------------------------------------------------------------------------
// Inline-asm f32 max: avoids llvm.maxnum canonicalization ops around WMMA /
// DPP results (v_max_num quiets NaNs itself, IEEE maximumNumber semantics).
// ---------------------------------------------------------------------------
__device__ inline float asm_maxf(float a, float b) {
    float d;
    asm("v_max_num_f32_e32 %0, %1, %2" : "=v"(d) : "v"(a), "v"(b));
    return d;
}

// ---------------------------------------------------------------------------
// Single-instruction DPP16 row-reduction steps: v_max/add_f32 with the DPP
// rotate applied to src0 — avoids the mov_dpp + canonicalize sequence.
// ---------------------------------------------------------------------------
#define ROW_MAX_STEP(v, n)                                                   \
    asm("v_max_num_f32_dpp %0, %1, %1 row_ror:" #n                           \
        " row_mask:0xf bank_mask:0xf bound_ctrl:1"                           \
        : "=v"(v) : "v"(v))
#define ROW_ADD_STEP(v, n)                                                   \
    asm("v_add_f32_dpp %0, %1, %1 row_ror:" #n                               \
        " row_mask:0xf bank_mask:0xf bound_ctrl:1"                           \
        : "=v"(v) : "v"(v))

__device__ inline float row_max16(float v) {
    ROW_MAX_STEP(v, 1); ROW_MAX_STEP(v, 2);
    ROW_MAX_STEP(v, 4); ROW_MAX_STEP(v, 8);
    return v;
}
__device__ inline float row_sum16(float v) {
    ROW_ADD_STEP(v, 1); ROW_ADD_STEP(v, 2);
    ROW_ADD_STEP(v, 4); ROW_ADD_STEP(v, 8);
    return v;
}

// ---------------------------------------------------------------------------
// Fragment loaders (CDNA5 wave32 WMMA layouts, CDNA5 ISA §7.12.2)
// ---------------------------------------------------------------------------

// A-matrix 16x32 bf16 from row-major bf16 (LDS or global).
__device__ inline v16bf load_frag_a(const bf16_t* __restrict__ base, int ld, int lane) {
    const int row  = lane & 15;
    const int half = lane >> 4;
    const bf16_t* p = base + row * ld + half * 8;
    union { v16bf v; unsigned int u[8]; } f;
#pragma unroll
    for (int pp = 0; pp < 8; ++pp) {
        const int k = 2 * (pp & 3) + (pp >> 2) * 16;   // 0,2,4,6,16,18,20,22
        f.u[pp] = *(const unsigned int*)(p + k);
    }
    return f.v;
}

// B-matrix 32x16 bf16 from a TRANSPOSED tile T[n][k]: packed pair reads (table k order).
__device__ inline v16bf load_frag_b_T(const bf16_t* __restrict__ T, int ld,
                                      int n_off, int lane) {
    const int n    = n_off + (lane & 15);
    const int half = lane >> 4;
    const bf16_t* p = T + n * ld + half * 8;
    union { v16bf v; unsigned int u[8]; } f;
#pragma unroll
    for (int pp = 0; pp < 8; ++pp)
        f.u[pp] = *(const unsigned int*)(p + 2 * (pp & 3) + (pp >> 2) * 16);
    return f.v;
}

// ---------------------------------------------------------------------------
// GEMM: Out[M,N] = (A[M,K] @ W[K,N] (+bias)) * oscale.
// Block 256 thr = 8 waves, tile 64x128; wave tile 64x16 (4 acc / 1 B frag).
// Double-buffered LDS, single barrier per K-step, async/reg prefetch.
// ---------------------------------------------------------------------------
template <typename AT, typename OT, bool HAS_BIAS>
__global__ __launch_bounds__(256)
void gemm_wmma_kernel(const AT* __restrict__ A, const float* __restrict__ W,
                      const float* __restrict__ bias, OT* __restrict__ Out,
                      int M, int N, int K, float oscale) {
    constexpr int LDA = 40;
    constexpr int LDW = 40;
    __shared__ bf16_t Asl[2][64 * LDA];    // [64 m][32 k]
    __shared__ bf16_t Wsl[2][128 * LDW];   // TRANSPOSED [128 n][32 k]

    const int lane = threadIdx.x & 31;
    const int wave = threadIdx.x >> 5;
    const int m0  = blockIdx.y * 64;
    const int n0b = blockIdx.x * 128;
    const int col = lane & 15, half = lane >> 4;

    // staging roles
    const int ar = threadIdx.x >> 2;            // A row 0..63
    const int ac = (threadIdx.x & 3) * 8;       // A col 0,8,16,24
    const int kp = threadIdx.x >> 4;            // W k-pair 0..15
    const int ng = (threadIdx.x & 15) * 8;      // W n-group

    const AT*    aptr = A + (long)(m0 + ar) * K + ac;
    const float* wptr = W + (long)(2 * kp) * N + n0b + ng;

    v8f acc[4] = {};
    if constexpr (HAS_BIAS) {
        const float bv = bias[n0b + wave * 16 + col];
#pragma unroll
        for (int mt = 0; mt < 4; ++mt)
#pragma unroll
            for (int r = 0; r < 8; ++r) acc[mt][r] = bv;
    }

    // ---- prologue: stage tile 0
    float areg[8];
    float w0r[8], w1r[8];
    if constexpr (sizeof(AT) == 2) {
#if HAVE_ASYNC
        ASYNC_LDS_B128(aptr, &Asl[0][ar * LDA + ac]);
#else
        *(uint4*)&Asl[0][ar * LDA + ac] = *(const uint4*)aptr;
#endif
    } else {
#pragma unroll
        for (int i = 0; i < 8; ++i) areg[i] = ((const float*)aptr)[i];
    }
#pragma unroll
    for (int j = 0; j < 8; ++j) { w0r[j] = wptr[j]; w1r[j] = wptr[N + j]; }

    const int KT = K / 32;
    for (int it = 0; it < KT; ++it) {
        const int p = it & 1;
        // ---- commit register-staged data into LDS buffer p
        if constexpr (sizeof(AT) != 2) {
            union { bf16_t h[8]; uint4 v; } t;
#pragma unroll
            for (int i = 0; i < 8; ++i) t.h[i] = (bf16_t)areg[i];
            *(uint4*)&Asl[p][ar * LDA + ac] = t.v;
        }
#pragma unroll
        for (int j = 0; j < 8; ++j) {
            union { bf16_t h[2]; unsigned int u; } pk;
            pk.h[0] = (bf16_t)w0r[j];
            pk.h[1] = (bf16_t)w1r[j];
            *(unsigned int*)&Wsl[p][(ng + j) * LDW + 2 * kp] = pk.u;
        }
        if constexpr (sizeof(AT) == 2) { WAIT_ASYNC(); }
        __syncthreads();

        // ---- prefetch tile it+1 (overlaps WMMAs below)
        if (it + 1 < KT) {
            aptr += 32;
            wptr += (long)32 * N;
            if constexpr (sizeof(AT) == 2) {
#if HAVE_ASYNC
                ASYNC_LDS_B128(aptr, &Asl[p ^ 1][ar * LDA + ac]);
#else
                *(uint4*)&Asl[p ^ 1][ar * LDA + ac] = *(const uint4*)aptr;
#endif
            } else {
#pragma unroll
                for (int i = 0; i < 8; ++i) areg[i] = ((const float*)aptr)[i];
            }
#pragma unroll
            for (int j = 0; j < 8; ++j) { w0r[j] = wptr[j]; w1r[j] = wptr[N + j]; }
        }

        // ---- compute on buffer p
        const v16bf b = load_frag_b_T(Wsl[p], LDW, wave * 16, lane);
#pragma unroll
        for (int mt = 0; mt < 4; ++mt) {
            const v16bf a = load_frag_a(&Asl[p][mt * 16 * LDA], LDA, lane);
            acc[mt] = __builtin_amdgcn_wmma_f32_16x16x32_bf16(
                false, a, false, b, (short)0, acc[mt], false, false);
        }
    }

#pragma unroll
    for (int mt = 0; mt < 4; ++mt)
#pragma unroll
        for (int r = 0; r < 8; ++r) {
            const int row = m0 + mt * 16 + r + 8 * half;
            Out[(long)row * N + n0b + wave * 16 + col] = (OT)(acc[mt][r] * oscale);
        }
}

// ---------------------------------------------------------------------------
// Flash attention: B=4, H=12, L=2048, D=64.  Q pre-scaled by 0.125*log2(e).
// grid = (L/128, B*H); block = 8 waves x 16 query rows.  Double-buffered
// K (async) and V (reg->transposed LDS) with one barrier per 32-key block.
// ---------------------------------------------------------------------------
__global__ __launch_bounds__(256)
void attn_wmma_kernel(const bf16_t* __restrict__ Q, const bf16_t* __restrict__ Km,
                      const bf16_t* __restrict__ V, bf16_t* __restrict__ Y) {
    constexpr int L = 2048, C = 768, H = 12, D = 64;
    constexpr int LDK  = 72;  // K tile [32 key][64 d]
    constexpr int LDVT = 40;  // V tile TRANSPOSED [64 d][32 key]
    constexpr int LDP  = 40;  // P tile [16 q][32 key] per wave

    __shared__ bf16_t Ksl[2][32 * LDK];
    __shared__ bf16_t Vsl[2][64 * LDVT];
    __shared__ bf16_t Psl[8 * 16 * LDP];

    const int lane = threadIdx.x & 31;
    const int wave = threadIdx.x >> 5;
    const int bh = blockIdx.y;
    const int b = bh / H, h = bh % H;
    const int q0 = blockIdx.x * 128 + wave * 16;
    const int col = lane & 15, half = lane >> 4;

    const long qbase = ((long)b * L + q0) * C + h * D;
    const v16bf aq0 = load_frag_a(Q + qbase, C, lane);        // d = 0..31
    const v16bf aq1 = load_frag_a(Q + qbase + 32, C, lane);   // d = 32..63

    v8f o0 = {}, o1 = {}, o2 = {}, o3 = {};
    float m_r[8], l_r[8];
#pragma unroll
    for (int r = 0; r < 8; ++r) { m_r[r] = -1e30f; l_r[r] = 0.0f; }

    bf16_t* Pw = Psl + wave * 16 * LDP;

    // staging roles
    const int kr = threadIdx.x >> 3;        // key row 0..31
    const int c8 = (threadIdx.x & 7) * 8;   // d offset
    const bf16_t* kptr = Km + ((long)b * L + kr) * C + h * D + c8;
    const bf16_t* vptr = V  + ((long)b * L + kr) * C + h * D + c8;

    // ---- prologue: stage block 0
#if HAVE_ASYNC
    ASYNC_LDS_B128(kptr, &Ksl[0][kr * LDK + c8]);
#else
    *(uint4*)&Ksl[0][kr * LDK + c8] = *(const uint4*)kptr;
#endif
    union vu_t { uint4 v; bf16_t e[8]; };
    vu_t vreg;
    vreg.v = *(const uint4*)vptr;

    constexpr int JT = L / 32;              // 64 key blocks
    for (int it = 0; it < JT; ++it) {
        const int p = it & 1;
        // ---- commit V regs transposed into buffer p
#pragma unroll
        for (int i = 0; i < 8; ++i)
            Vsl[p][(c8 + i) * LDVT + kr] = vreg.e[i];
        WAIT_ASYNC();
        __syncthreads();

        // ---- prefetch block it+1 (overlaps compute below)
        if (it + 1 < JT) {
            kptr += (long)32 * C;
            vptr += (long)32 * C;
#if HAVE_ASYNC
            ASYNC_LDS_B128(kptr, &Ksl[p ^ 1][kr * LDK + c8]);
#else
            *(uint4*)&Ksl[p ^ 1][kr * LDK + c8] = *(const uint4*)kptr;
#endif
            vreg.v = *(const uint4*)vptr;
        }

        // ---- scores S[16q x 32k] (log2 domain; scale folded into Q)
        v8f s0 = {}, s1 = {};
        s0 = __builtin_amdgcn_wmma_f32_16x16x32_bf16(false, aq0, false,
                 load_frag_b_T(Ksl[p], LDK, 0,  lane), (short)0, s0, false, false);
        s0 = __builtin_amdgcn_wmma_f32_16x16x32_bf16(false, aq1, false,
                 load_frag_b_T(Ksl[p] + 32, LDK, 0,  lane), (short)0, s0, false, false);
        s1 = __builtin_amdgcn_wmma_f32_16x16x32_bf16(false, aq0, false,
                 load_frag_b_T(Ksl[p], LDK, 16, lane), (short)0, s1, false, false);
        s1 = __builtin_amdgcn_wmma_f32_16x16x32_bf16(false, aq1, false,
                 load_frag_b_T(Ksl[p] + 32, LDK, 16, lane), (short)0, s1, false, false);

        // ---- online softmax (all maxes via v_max_num asm: no canonicalize)
#pragma unroll
        for (int r = 0; r < 8; ++r) {
            const float sc0 = s0[r];
            const float sc1 = s1[r];
            const float mx = row_max16(asm_maxf(sc0, sc1));
            const float m_new = asm_maxf(m_r[r], mx);
            const float p0 = exp2f(sc0 - m_new);
            const float p1 = exp2f(sc1 - m_new);
            const float alpha = exp2f(m_r[r] - m_new);
            const float rs = row_sum16(p0 + p1);
            l_r[r] = l_r[r] * alpha + rs;
            m_r[r] = m_new;
            o0[r] *= alpha; o1[r] *= alpha; o2[r] *= alpha; o3[r] *= alpha;
            const int prow = r + 8 * half;
            Pw[prow * LDP + col]      = (bf16_t)p0;
            Pw[prow * LDP + 16 + col] = (bf16_t)p1;
        }

        // ---- O += P[16x32] @ V[32x64]
        const v16bf ap = load_frag_a(Pw, LDP, lane);
        o0 = __builtin_amdgcn_wmma_f32_16x16x32_bf16(false, ap, false,
                 load_frag_b_T(Vsl[p], LDVT, 0,  lane), (short)0, o0, false, false);
        o1 = __builtin_amdgcn_wmma_f32_16x16x32_bf16(false, ap, false,
                 load_frag_b_T(Vsl[p], LDVT, 16, lane), (short)0, o1, false, false);
        o2 = __builtin_amdgcn_wmma_f32_16x16x32_bf16(false, ap, false,
                 load_frag_b_T(Vsl[p], LDVT, 32, lane), (short)0, o2, false, false);
        o3 = __builtin_amdgcn_wmma_f32_16x16x32_bf16(false, ap, false,
                 load_frag_b_T(Vsl[p], LDVT, 48, lane), (short)0, o3, false, false);
    }

    // ---- normalize and store y (bf16, merged-head layout [B,L,C])
#pragma unroll
    for (int r = 0; r < 8; ++r) {
        const float inv = 1.0f / l_r[r];
        const int row = q0 + r + 8 * half;
        const long yb = ((long)b * L + row) * C + h * D + col;
        Y[yb]      = (bf16_t)(o0[r] * inv);
        Y[yb + 16] = (bf16_t)(o1[r] * inv);
        Y[yb + 32] = (bf16_t)(o2[r] * inv);
        Y[yb + 48] = (bf16_t)(o3[r] * inv);
    }
}

// ---------------------------------------------------------------------------
// Launch
// ---------------------------------------------------------------------------
extern "C" void kernel_launch(void* const* d_in, const int* in_sizes, int n_in,
                              void* d_out, int out_size, void* d_ws, size_t ws_size,
                              hipStream_t stream) {
    constexpr int B = 4, L = 2048, C = 768;
    constexpr int M = B * L;             // 8192
    constexpr long ML = (long)M * C;

    const float* h_private = (const float*)d_in[0];
    const float* h_shared  = (const float*)d_in[1];
    const float* Wq  = (const float*)d_in[2];
    const float* bq  = (const float*)d_in[3];
    const float* Wk  = (const float*)d_in[4];
    const float* bk  = (const float*)d_in[5];
    const float* Wv  = (const float*)d_in[6];
    const float* bv  = (const float*)d_in[7];
    const float* Wvt = (const float*)d_in[8];
    const float* Wcp = (const float*)d_in[9];
    float* out = (float*)d_out;

    bf16_t* qb = (bf16_t*)d_ws;
    bf16_t* kb = qb + ML;
    bf16_t* vb = kb + ML;
    bf16_t* tb = vb + ML;   // value_mlp intermediate
    bf16_t* yb = tb + ML;   // attention output (pre c_proj)

    const dim3 blk(256);
    const dim3 gg(C / 128, M / 64);      // (6, 128)

    // softmax scale (1/sqrt(64)) * log2(e) folded into Q projection
    const float qscale = 0.125f * 1.44269504088896340736f;

    gemm_wmma_kernel<float,  bf16_t, true ><<<gg, blk, 0, stream>>>(h_shared,  Wq,  bq, qb, M, C, C, qscale);
    gemm_wmma_kernel<float,  bf16_t, true ><<<gg, blk, 0, stream>>>(h_shared,  Wk,  bk, kb, M, C, C, 1.0f);
    gemm_wmma_kernel<float,  bf16_t, true ><<<gg, blk, 0, stream>>>(h_private, Wv,  bv, tb, M, C, C, 1.0f);
    gemm_wmma_kernel<bf16_t, bf16_t, false><<<gg, blk, 0, stream>>>(tb,        Wvt, nullptr, vb, M, C, C, 1.0f);

    attn_wmma_kernel<<<dim3(L / 128, 48), blk, 0, stream>>>(qb, kb, vb, yb);

    gemm_wmma_kernel<bf16_t, float, false><<<gg, blk, 0, stream>>>(yb, Wcp, nullptr, out, M, C, C, 1.0f);
}
